// TemporalPCN_17918603559172
// MI455X (gfx1250) — matrix operand order repaired and checked by reading, not compiled, not run on hardware
//
#include <hip/hip_runtime.h>
#include <hip/hip_bf16.h>
#include <math.h>

// ---------------------------------------------------------------------------
// TemporalPCN on MI455X (gfx1250, wave32, WMMA)
//   g = tanh(prev_z @ Wr^T + v @ Win^T);  z = g
//   repeat inf_iters:
//     pred = tanh(z @ Wout^T); t = (1-pred^2)*(p-pred)
//     z   -= 0.05 * ((z - g) - t @ Wout + 1e-4*sign(z))
// Compute-bound (~740 GFLOP, data ~100MB << 192MB L2) -> bf16 WMMA, f32 accum.
// ---------------------------------------------------------------------------

typedef __attribute__((ext_vector_type(16))) __bf16 v16bf;
typedef __attribute__((ext_vector_type(8)))  float  v8f;

#define BM 128
#define BN 128
#define BK 32
#define SAK 40      // padded LDS row stride (bf16 elems): avoids ds bank conflicts
#define SBK 40
#define NTHREADS 256

#define B_DIM  4096
#define NG     2048
#define NV     128
#define NP     1024
#define MAX_ITERS 20

union FragU { uint4 q[2]; v16bf v; };

// K-loop of a 128x128 tile GEMM:  C += A[rowBase.., k] * Bm[colBase.., k]^T
// A: [M][lda] bf16 row-major (K contiguous).  Bm: [N][ldb] bf16 (K contiguous),
// i.e. Bm holds B^T so every wmma B-fragment read is K-contiguous.
__device__ __forceinline__ void wmma_phase(
    const __bf16* __restrict__ A, int lda,
    const __bf16* __restrict__ Bm, int ldb,
    int K, int rowBase, int colBase,
    __bf16* sA, __bf16* sB,
    int tid, int lane, int wm, int wn,
    v8f acc[2][4])
{
    const int kh  = (lane & 16) ? 8  : 0;   // A frag: lanes 16-31 hold K+8 half
    const int klo = (lane & 16) ? 8  : 0;   // B frag: lanes 16-31 hold K+8 half
    const int ln  = lane & 15;

    for (int k0 = 0; k0 < K; k0 += BK) {
        __syncthreads();
        // stage 128x32 bf16 tiles of A and B^T into LDS (16B chunks, 2 per thread)
#pragma unroll
        for (int t = 0; t < 2; ++t) {
            int chunk = tid + t * NTHREADS;     // 0..511
            int row   = chunk >> 2;             // 0..127
            int kc    = (chunk & 3) << 3;       // 0,8,16,24
            *(uint4*)&sA[row * SAK + kc] =
                *(const uint4*)&A[(size_t)(rowBase + row) * lda + k0 + kc];
            *(uint4*)&sB[row * SBK + kc] =
                *(const uint4*)&Bm[(size_t)(colBase + row) * ldb + k0 + kc];
        }
        __syncthreads();

        FragU a[2], b[4];
#pragma unroll
        for (int tm = 0; tm < 2; ++tm) {
            int r = wm * 32 + tm * 16 + ln;
            a[tm].q[0] = *(const uint4*)&sA[r * SAK + kh];        // K = kh..kh+7
            a[tm].q[1] = *(const uint4*)&sA[r * SAK + 16 + kh];   // K = 16+kh..
        }
#pragma unroll
        for (int tn = 0; tn < 4; ++tn) {
            int c = wn * 64 + tn * 16 + ln;
            b[tn].q[0] = *(const uint4*)&sB[c * SBK + klo];       // K = klo..klo+7
            b[tn].q[1] = *(const uint4*)&sB[c * SBK + klo + 16];  // K = 16+klo..
        }
#pragma unroll
        for (int tm = 0; tm < 2; ++tm)
#pragma unroll
            for (int tn = 0; tn < 4; ++tn)
                acc[tm][tn] = __builtin_amdgcn_wmma_f32_16x16x32_bf16(
                    false, a[tm].v, false, b[tn].v,
                    (short)0, acc[tm][tn], false, false);
    }
}

// ---- GEMM-G: g = tanh(prev_z@Wr^T + v@Win^T); z=g (f32 + bf16 copies) -------
__global__ void __launch_bounds__(NTHREADS)
gemm_g_kernel(const __bf16* __restrict__ pz,  const __bf16* __restrict__ wr,
              const __bf16* __restrict__ vb,  const __bf16* __restrict__ win,
              float* __restrict__ g, float* __restrict__ z,
              __bf16* __restrict__ zbf)
{
    __shared__ __bf16 sA[BM * SAK];
    __shared__ __bf16 sB[BN * SBK];
    const int tid  = threadIdx.x;
    const int lane = tid & 31, wid = tid >> 5;
    const int wm = wid & 3, wn = wid >> 2;
    const int ln = lane & 15, rhalf = (lane >> 4) << 3;
    const int rowBase = blockIdx.y * BM, colBase = blockIdx.x * BN;

    const v8f vzero = {0.f,0.f,0.f,0.f,0.f,0.f,0.f,0.f};
    v8f acc[2][4];
#pragma unroll
    for (int i = 0; i < 2; ++i)
#pragma unroll
        for (int j = 0; j < 4; ++j) acc[i][j] = vzero;

    wmma_phase(pz, NG, wr,  NG, NG, rowBase, colBase, sA, sB, tid, lane, wm, wn, acc);
    wmma_phase(vb, NV, win, NV, NV, rowBase, colBase, sA, sB, tid, lane, wm, wn, acc);

#pragma unroll
    for (int tm = 0; tm < 2; ++tm)
#pragma unroll
        for (int tn = 0; tn < 4; ++tn)
#pragma unroll
            for (int r = 0; r < 8; ++r) {
                int row = rowBase + wm * 32 + tm * 16 + rhalf + r;
                int col = colBase + wn * 64 + tn * 16 + ln;
                size_t idx = (size_t)row * NG + col;
                float gv = tanhf(acc[tm][tn][r]);
                g[idx]   = gv;
                z[idx]   = gv;
                zbf[idx] = (__bf16)gv;
            }
}

// ---- GEMM-F: t = (1-tanh^2(z@Wout^T)) * (p - tanh(z@Wout^T)) ---------------
__global__ void __launch_bounds__(NTHREADS)
gemm_f_kernel(const __bf16* __restrict__ zbf, const __bf16* __restrict__ wout,
              const float* __restrict__ p, __bf16* __restrict__ tbf,
              const int* __restrict__ inf_iters, int iter)
{
    if (iter >= *inf_iters) return;   // uniform grid-wide early exit
    __shared__ __bf16 sA[BM * SAK];
    __shared__ __bf16 sB[BN * SBK];
    const int tid  = threadIdx.x;
    const int lane = tid & 31, wid = tid >> 5;
    const int wm = wid & 3, wn = wid >> 2;
    const int ln = lane & 15, rhalf = (lane >> 4) << 3;
    const int rowBase = blockIdx.y * BM, colBase = blockIdx.x * BN;

    const v8f vzero = {0.f,0.f,0.f,0.f,0.f,0.f,0.f,0.f};
    v8f acc[2][4];
#pragma unroll
    for (int i = 0; i < 2; ++i)
#pragma unroll
        for (int j = 0; j < 4; ++j) acc[i][j] = vzero;

    wmma_phase(zbf, NG, wout, NG, NG, rowBase, colBase, sA, sB, tid, lane, wm, wn, acc);

#pragma unroll
    for (int tm = 0; tm < 2; ++tm)
#pragma unroll
        for (int tn = 0; tn < 4; ++tn)
#pragma unroll
            for (int r = 0; r < 8; ++r) {
                int row = rowBase + wm * 32 + tm * 16 + rhalf + r;
                int col = colBase + wn * 64 + tn * 16 + ln;
                size_t idx = (size_t)row * NP + col;
                float pred = tanhf(acc[tm][tn][r]);
                float t    = (1.0f - pred * pred) * (p[idx] - pred);
                tbf[idx]   = (__bf16)t;
            }
}

// ---- GEMM-B: z -= 0.05*((z-g) - t@Wout + 1e-4*sign(z)) ---------------------
__global__ void __launch_bounds__(NTHREADS)
gemm_b_kernel(const __bf16* __restrict__ tbf, const __bf16* __restrict__ wot,
              const float* __restrict__ g, float* __restrict__ z,
              __bf16* __restrict__ zbf,
              const int* __restrict__ inf_iters, int iter)
{
    if (iter >= *inf_iters) return;
    __shared__ __bf16 sA[BM * SAK];
    __shared__ __bf16 sB[BN * SBK];
    const int tid  = threadIdx.x;
    const int lane = tid & 31, wid = tid >> 5;
    const int wm = wid & 3, wn = wid >> 2;
    const int ln = lane & 15, rhalf = (lane >> 4) << 3;
    const int rowBase = blockIdx.y * BM, colBase = blockIdx.x * BN;

    const v8f vzero = {0.f,0.f,0.f,0.f,0.f,0.f,0.f,0.f};
    v8f acc[2][4];
#pragma unroll
    for (int i = 0; i < 2; ++i)
#pragma unroll
        for (int j = 0; j < 4; ++j) acc[i][j] = vzero;

    wmma_phase(tbf, NP, wot, NP, NP, rowBase, colBase, sA, sB, tid, lane, wm, wn, acc);

#pragma unroll
    for (int tm = 0; tm < 2; ++tm)
#pragma unroll
        for (int tn = 0; tn < 4; ++tn)
#pragma unroll
            for (int r = 0; r < 8; ++r) {
                int row = rowBase + wm * 32 + tm * 16 + rhalf + r;
                int col = colBase + wn * 64 + tn * 16 + ln;
                size_t idx = (size_t)row * NG + col;
                float corr = acc[tm][tn][r];
                float zc = z[idx], gv = g[idx];
                float sgn = (zc > 0.f) ? 1.f : ((zc < 0.f) ? -1.f : 0.f);
                float zn = zc - 0.05f * ((zc - gv) - corr + 1e-4f * sgn);
                z[idx]   = zn;
                zbf[idx] = (__bf16)zn;
            }
}

// ---- conversion helpers ----------------------------------------------------
__global__ void cvt_bf16(const float* __restrict__ s, __bf16* __restrict__ d, int n) {
    int i = blockIdx.x * blockDim.x + threadIdx.x;
    if (i < n) d[i] = (__bf16)s[i];
}
// d[c*R + r] = s[r*C + c]  (store transposed, bf16)
__global__ void cvt_t_bf16(const float* __restrict__ s, __bf16* __restrict__ d,
                           int R, int C) {
    long i = (long)blockIdx.x * blockDim.x + threadIdx.x;
    if (i >= (long)R * C) return;
    int r = (int)(i % R), c = (int)(i / R);
    d[i] = (__bf16)s[(size_t)r * C + c];
}

// ---------------------------------------------------------------------------
extern "C" void kernel_launch(void* const* d_in, const int* in_sizes, int n_in,
                              void* d_out, int out_size, void* d_ws, size_t ws_size,
                              hipStream_t stream) {
    const float* v      = (const float*)d_in[0];   // [4096,128]
    const float* prev_z = (const float*)d_in[1];   // [4096,2048]
    const float* p      = (const float*)d_in[2];   // [4096,1024]
    const float* Wr     = (const float*)d_in[3];   // [2048,2048]
    const float* Win    = (const float*)d_in[4];   // [2048,128]
    const float* Wout   = (const float*)d_in[5];   // [1024,2048]
    const int* inf_it   = (const int*)d_in[6];
    float* zout = (float*)d_out;                   // z lives here (f32)

    char* ws = (char*)d_ws;
    size_t off = 0;
    auto carve = [&](size_t bytes) {
        void* ptr = ws + off;
        off += (bytes + 255) & ~(size_t)255;
        return ptr;
    };
    __bf16* WrB   = (__bf16*)carve((size_t)NG * NG * 2);   // [Ng][Ng]
    __bf16* WinB  = (__bf16*)carve((size_t)NG * NV * 2);   // [Ng][Nv]
    __bf16* WoutB = (__bf16*)carve((size_t)NP * NG * 2);   // [Np][Ng]
    __bf16* WoTB  = (__bf16*)carve((size_t)NG * NP * 2);   // [Ng][Np] = Wout^T
    __bf16* PzB   = (__bf16*)carve((size_t)B_DIM * NG * 2);
    __bf16* VB    = (__bf16*)carve((size_t)B_DIM * NV * 2);
    float*  G     = (float*) carve((size_t)B_DIM * NG * 4);
    __bf16* ZB    = (__bf16*)carve((size_t)B_DIM * NG * 2);
    __bf16* TB    = (__bf16*)carve((size_t)B_DIM * NP * 2);
    // total workspace ~94 MB

    // one-time bf16 conversions (L2-resident afterwards)
    cvt_bf16<<<(NG*NG + 255) / 256, 256, 0, stream>>>(Wr, WrB, NG * NG);
    cvt_bf16<<<(NG*NV + 255) / 256, 256, 0, stream>>>(Win, WinB, NG * NV);
    cvt_bf16<<<(NP*NG + 255) / 256, 256, 0, stream>>>(Wout, WoutB, NP * NG);
    cvt_t_bf16<<<(NG*NP + 255) / 256, 256, 0, stream>>>(Wout, WoTB, NP, NG);
    cvt_bf16<<<(B_DIM*NG + 255) / 256, 256, 0, stream>>>(prev_z, PzB, B_DIM * NG);
    cvt_bf16<<<(B_DIM*NV + 255) / 256, 256, 0, stream>>>(v, VB, B_DIM * NV);

    // g = z0 = tanh(prev_z@Wr^T + v@Win^T)
    gemm_g_kernel<<<dim3(NG / BN, B_DIM / BM), NTHREADS, 0, stream>>>(
        PzB, WrB, VB, WinB, G, zout, ZB);

    // 20 relaxation iterations (each kernel self-guards on *inf_iters)
    for (int it = 0; it < MAX_ITERS; ++it) {
        gemm_f_kernel<<<dim3(NP / BN, B_DIM / BM), NTHREADS, 0, stream>>>(
            ZB, WoutB, p, TB, inf_it, it);
        gemm_b_kernel<<<dim3(NG / BN, B_DIM / BM), NTHREADS, 0, stream>>>(
            TB, WoTB, G, zout, ZB, inf_it, it);
    }
}
